// PatchCoreAnomalyHead_28991029248665
// MI455X (gfx1250) — matrix operand order repaired and checked by reading, not compiled, not run on hardware
//
#include <hip/hip_runtime.h>
#include <hip/hip_bf16.h>

typedef float  v2f   __attribute__((ext_vector_type(2)));
typedef float  v8f   __attribute__((ext_vector_type(8)));
typedef __bf16 v8bf  __attribute__((ext_vector_type(8)));
typedef __bf16 v16bf __attribute__((ext_vector_type(16)));

#define NROWS 32768   // B*L
#define C1    1024
#define C2    512
#define C3    256
#define MBANK 16384

// ---------------------------------------------------------------------------
// K0: memory bank prep: fp32 -> bf16 copy + m2[row] = sum(bf16(mb)^2) in fp32.
// One wave per bank row; 8 rows per 256-thread block.
// ---------------------------------------------------------------------------
__global__ __launch_bounds__(256) void k_mbprep(const float* __restrict__ MB,
                                                __bf16* __restrict__ MBb,
                                                float* __restrict__ m2) {
  const int lane = threadIdx.x & 31;
  const int wave = threadIdx.x >> 5;
  const int row  = blockIdx.x * 8 + wave;
  const float* src = MB + (size_t)row * C3 + lane * 8;
  float4 f0 = *(const float4*)(src);
  float4 f1 = *(const float4*)(src + 4);
  float in[8] = {f0.x, f0.y, f0.z, f0.w, f1.x, f1.y, f1.z, f1.w};
  v8bf pack;
  float ss = 0.f;
#pragma unroll
  for (int i = 0; i < 8; ++i) {
    __bf16 b = (__bf16)in[i];
    pack[i] = b;
    float v = (float)b;
    ss += v * v;
  }
  *(v8bf*)(MBb + (size_t)row * C3 + lane * 8) = pack;
#pragma unroll
  for (int m = 16; m > 0; m >>= 1) ss += __shfl_xor(ss, m, 32);
  if (lane == 0) m2[row] = ss;
}

// ---------------------------------------------------------------------------
// K1: H = relu(F @ W1 + b1), fp32 exact via V_WMMA_F32_16X16X4_F32.
// One wave -> one 16x16 output tile; 4 waves/block cover 64 columns.
// ---------------------------------------------------------------------------
__global__ __launch_bounds__(128) void k_gemm1_relu(const float* __restrict__ F,
                                                    const float* __restrict__ W1,
                                                    const float* __restrict__ b1,
                                                    float* __restrict__ H) {
  const int lane = threadIdx.x & 31;
  const int wave = threadIdx.x >> 5;
  const int n    = lane & 15;       // A row (M) and B/C column (N)
  const int h    = lane >> 4;       // lane-half selects K pair / M half
  const int row0 = blockIdx.x * 16;
  const int col  = (blockIdx.y * 4 + wave) * 16 + n;
  const float* arow = F + (size_t)(row0 + n) * C1;
  v8f c = {};
#pragma unroll 4
  for (int k = 0; k < C1; k += 4) {
    v2f a = *(const v2f*)(arow + k + 2 * h);          // A[m][k+2h .. +1]
    v2f b;
    b.x = W1[(size_t)(k + 2 * h) * C2 + col];         // B[k+2h][n]
    b.y = W1[(size_t)(k + 2 * h + 1) * C2 + col];     // B[k+2h+1][n]
    c = __builtin_amdgcn_wmma_f32_16x16x4_f32(false, a, false, b, (short)0, c,
                                              false, false);
  }
  const float bias = b1[col];
#pragma unroll
  for (int r = 0; r < 8; ++r) {
    const int row = row0 + r + 8 * h;
    float v = c[r] + bias;
    H[(size_t)row * C2 + col] = v > 0.f ? v : 0.f;
  }
}

// ---------------------------------------------------------------------------
// K2: P = H @ W2 + b2 (fp32 WMMA), written directly as bf16.
// ---------------------------------------------------------------------------
__global__ __launch_bounds__(128) void k_gemm2_bf16(const float* __restrict__ H,
                                                    const float* __restrict__ W2,
                                                    const float* __restrict__ b2,
                                                    __bf16* __restrict__ Pb) {
  const int lane = threadIdx.x & 31;
  const int wave = threadIdx.x >> 5;
  const int n    = lane & 15;
  const int h    = lane >> 4;
  const int row0 = blockIdx.x * 16;
  const int col  = (blockIdx.y * 4 + wave) * 16 + n;
  const float* arow = H + (size_t)(row0 + n) * C2;
  v8f c = {};
#pragma unroll 4
  for (int k = 0; k < C2; k += 4) {
    v2f a = *(const v2f*)(arow + k + 2 * h);
    v2f b;
    b.x = W2[(size_t)(k + 2 * h) * C3 + col];
    b.y = W2[(size_t)(k + 2 * h + 1) * C3 + col];
    c = __builtin_amdgcn_wmma_f32_16x16x4_f32(false, a, false, b, (short)0, c,
                                              false, false);
  }
  const float bias = b2[col];
#pragma unroll
  for (int r = 0; r < 8; ++r) {
    const int row = row0 + r + 8 * h;
    Pb[(size_t)row * C3 + col] = (__bf16)(c[r] + bias);
  }
}

// ---------------------------------------------------------------------------
// K2b: x2[row] = sum over 256 cols of bf16(P)^2, accumulated in fp32.
// One wave per row.
// ---------------------------------------------------------------------------
__global__ __launch_bounds__(256) void k_x2(const __bf16* __restrict__ Pb,
                                            float* __restrict__ x2) {
  const int lane = threadIdx.x & 31;
  const int wave = threadIdx.x >> 5;
  const int row  = blockIdx.x * 8 + wave;
  v8bf v = *(const v8bf*)(Pb + (size_t)row * C3 + lane * 8);
  float ss = 0.f;
#pragma unroll
  for (int i = 0; i < 8; ++i) {
    float f = (float)v[i];
    ss += f * f;
  }
#pragma unroll
  for (int m = 16; m > 0; m >>= 1) ss += __shfl_xor(ss, m, 32);
  if (lane == 0) x2[row] = ss;
}

// ---------------------------------------------------------------------------
// K3: fused distance with 2-way register blocking.
// Each wave holds TWO 16x256 bf16 A tiles (128 VGPRs, loop invariant) and
// streams the bank in 16-row chunks: each 32-byte B block feeds 2 WMMAs
// (halves L2 B-traffic, provides 2 independent accumulator chains).
// Tracks qmin = min_cols(m2 - 2*xm); x2 + clamp + sqrt folded in at the end
// (valid since x2 is constant per row and non-negative).
// ---------------------------------------------------------------------------
__global__ __launch_bounds__(128) void k_dist(const __bf16* __restrict__ Pb,
                                              const __bf16* __restrict__ MBb,
                                              const float* __restrict__ x2,
                                              const float* __restrict__ m2,
                                              float* __restrict__ out) {
  const int lane  = threadIdx.x & 31;
  const int wave  = threadIdx.x >> 5;
  const int n     = lane & 15;
  const int h     = lane >> 4;
  const int tile0 = (blockIdx.x * 4 + wave) * 2;
  const int row0  = tile0 * 16;        // first tile rows [row0, row0+16)
  const int row1  = row0 + 16;         // second tile rows

  // Preload both A tiles: 16 rows x 256 bf16 -> 8 K-blocks of v16bf each.
  // A k order used here (contiguous, same as B): lanes 0-15: K = j*32 + [0..15];
  //                      lanes 16-31: K = j*32 + [16..31].
  v16bf A0[8], A1[8];
  const __bf16* prow0 = Pb + (size_t)(row0 + n) * C3;
  const __bf16* prow1 = Pb + (size_t)(row1 + n) * C3;
#pragma unroll
  for (int j = 0; j < 8; ++j) {
    v8bf lo0 = *(const v8bf*)(prow0 + j * 32 + 16 * h);
    v8bf hi0 = *(const v8bf*)(prow0 + j * 32 + 16 * h + 8);
    A0[j] = __builtin_shufflevector(lo0, hi0, 0, 1, 2, 3, 4, 5, 6, 7, 8, 9, 10,
                                    11, 12, 13, 14, 15);
    v8bf lo1 = *(const v8bf*)(prow1 + j * 32 + 16 * h);
    v8bf hi1 = *(const v8bf*)(prow1 + j * 32 + 16 * h + 8);
    A1[j] = __builtin_shufflevector(lo1, hi1, 0, 1, 2, 3, 4, 5, 6, 7, 8, 9, 10,
                                    11, 12, 13, 14, 15);
  }

  float qmin0[8], qmin1[8];
#pragma unroll
  for (int r = 0; r < 8; ++r) {
    qmin0[r] = __builtin_inff();
    qmin1[r] = __builtin_inff();
  }

  for (int bank0 = 0; bank0 < MBANK; bank0 += 16) {
    const float m2c = m2[bank0 + n];
    // B k order used here (contiguous, same as A): lane = column (bank row); lanes 0-15 take the
    // first 16 K of each 32-K block, lanes 16-31 the second 16 -> one
    // contiguous 32-byte load per lane per K-block.
    const __bf16* brow = MBb + (size_t)(bank0 + n) * C3 + 16 * h;
    if (bank0 + 16 < MBANK) {  // prefetch the B row after this lane's (512 B/row)
      __builtin_prefetch(brow + C3, 0, 3);
      __builtin_prefetch(brow + C3 + 128, 0, 3);
    }
    v8f acc0 = {}, acc1 = {};
#pragma unroll
    for (int j = 0; j < 8; ++j) {
      v16bf Bj = *(const v16bf*)(brow + j * 32);
      acc0 = __builtin_amdgcn_wmma_f32_16x16x32_bf16(false, A0[j], false, Bj,
                                                     (short)0, acc0, false,
                                                     false);
      acc1 = __builtin_amdgcn_wmma_f32_16x16x32_bf16(false, A1[j], false, Bj,
                                                     (short)0, acc1, false,
                                                     false);
    }
#pragma unroll
    for (int r = 0; r < 8; ++r) {
      float q0 = __builtin_fmaf(-2.f, acc0[r], m2c);
      float q1 = __builtin_fmaf(-2.f, acc1[r], m2c);
      qmin0[r] = q0 < qmin0[r] ? q0 : qmin0[r];
      qmin1[r] = q1 < qmin1[r] ? q1 : qmin1[r];
    }
  }

  // Min over the 16 columns held by each lane-half, then fold in x2, clamp,
  // sqrt:  min_m max(x2 - 2xm + m2, 0) == max(x2 + min_m(m2 - 2xm), 0).
#pragma unroll
  for (int r = 0; r < 8; ++r) {
    float v0 = qmin0[r], v1 = qmin1[r];
    v0 = fminf(v0, __shfl_xor(v0, 1, 32));
    v1 = fminf(v1, __shfl_xor(v1, 1, 32));
    v0 = fminf(v0, __shfl_xor(v0, 2, 32));
    v1 = fminf(v1, __shfl_xor(v1, 2, 32));
    v0 = fminf(v0, __shfl_xor(v0, 4, 32));
    v1 = fminf(v1, __shfl_xor(v1, 4, 32));
    v0 = fminf(v0, __shfl_xor(v0, 8, 32));
    v1 = fminf(v1, __shfl_xor(v1, 8, 32));
    qmin0[r] = v0;
    qmin1[r] = v1;
  }
  if (n == 0) {  // lane 0 -> rows r+0..7, lane 16 -> rows r+8..15 (per tile)
#pragma unroll
    for (int r = 0; r < 8; ++r) {
      const int ra = row0 + r + 8 * h;
      const int rb = row1 + r + 8 * h;
      float d0 = x2[ra] + qmin0[r];
      float d1 = x2[rb] + qmin1[r];
      out[ra] = __builtin_sqrtf(d0 > 0.f ? d0 : 0.f);
      out[rb] = __builtin_sqrtf(d1 > 0.f ? d1 : 0.f);
    }
  }
}

// ---------------------------------------------------------------------------
extern "C" void kernel_launch(void* const* d_in, const int* in_sizes, int n_in,
                              void* d_out, int out_size, void* d_ws,
                              size_t ws_size, hipStream_t stream) {
  (void)in_sizes; (void)n_in; (void)out_size; (void)ws_size;
  const float* F  = (const float*)d_in[0];
  const float* W1 = (const float*)d_in[1];
  const float* b1 = (const float*)d_in[2];
  const float* W2 = (const float*)d_in[3];
  const float* b2 = (const float*)d_in[4];
  const float* MB = (const float*)d_in[5];
  float* out = (float*)d_out;

  char* ws = (char*)d_ws;
  float*  H   = (float*)(ws);                          // 32768*512*4  = 64 MB
  __bf16* Pb  = (__bf16*)(ws + (size_t)67108864);      // 32768*256*2  = 16 MB
  __bf16* MBb = (__bf16*)(ws + (size_t)83886080);      // 16384*256*2  =  8 MB
  float*  x2  = (float*)(ws + (size_t)92274688);       // 32768*4      = 128 KB
  float*  m2  = (float*)(ws + (size_t)92405760);       // 16384*4      = 64 KB

  k_mbprep<<<dim3(MBANK / 8), 256, 0, stream>>>(MB, MBb, m2);
  k_gemm1_relu<<<dim3(NROWS / 16, C2 / 64), 128, 0, stream>>>(F, W1, b1, H);
  k_gemm2_bf16<<<dim3(NROWS / 16, C3 / 64), 128, 0, stream>>>(H, W2, b2, Pb);
  k_x2<<<dim3(NROWS / 8), 256, 0, stream>>>(Pb, x2);
  // 2048 row tiles, 2 per wave, 4 waves per block -> 256 blocks.
  k_dist<<<dim3(NROWS / 16 / 8), 128, 0, stream>>>(Pb, MBb, x2, m2, out);
}
